// BertSelfAttention_2894807957924
// MI455X (gfx1250) — matrix operand order repaired and checked by reading, not compiled, not run on hardware
//
#include <hip/hip_runtime.h>
#include <hip/hip_bf16.h>

typedef _Float16 v4h  __attribute__((ext_vector_type(4)));
typedef _Float16 v8h  __attribute__((ext_vector_type(8)));
typedef _Float16 v16h __attribute__((ext_vector_type(16)));
typedef float    v8f  __attribute__((ext_vector_type(8)));

#define DEVINL __device__ __forceinline__

constexpr int Bc  = 4;
constexpr int Sc  = 1024;
constexpr int HIDc = 1024;
constexpr int NHc = 16;
constexpr int HDc = 64;
constexpr int Mrows = Bc * Sc;   // 4096

DEVINL v16h cat8(v8h lo, v8h hi) {
  return __builtin_shufflevector(lo, hi, 0,1,2,3,4,5,6,7,8,9,10,11,12,13,14,15);
}

// A fragment 16x32 f16 from row-major [*, ld] at (row0, k0).
// ISA layout: lanes 0-15 hold rows, halves {g*8..g*8+7} and {16+g*8..16+g*8+7}, g=lane>>4.
DEVINL v16h load_a_frag(const _Float16* base, int ld, int row0, int k0, int lane) {
  int r = lane & 15, g = lane >> 4;
  const _Float16* p = base + (size_t)(row0 + r) * ld + k0 + g * 8;
  v8h lo = *(const v8h*)p;         // k = k0 + g*8 .. +7
  v8h hi = *(const v8h*)(p + 16);  // k = k0 + 16 + g*8 .. +7
  return cat8(lo, hi);
}

// B fragment 32x16 f16 where columns are contiguous in memory:
// baseT is [N][ld] row-major, row n of baseT == column n of B.
// Lane holds column (col0 + lane&15), k = k0 + g*8 .. +7 and k0 + 16 + g*8 .. +7.
DEVINL v16h load_b_frag(const _Float16* baseT, int ld, int col0, int k0, int lane) {
  int n = lane & 15, g = lane >> 4;
  const _Float16* p = baseT + (size_t)(col0 + n) * ld + k0 + g * 8;
  return cat8(*(const v8h*)p, *(const v8h*)(p + 16));  // two 16-byte loads
}

// ---------------------------------------------------------------- cast f32->f16
__global__ __launch_bounds__(256) void cast_f32_f16_kernel(
    const float* __restrict__ src, _Float16* __restrict__ dst, int n4) {
  int i = blockIdx.x * blockDim.x + threadIdx.x;
  if (i < n4) {
    float4 v = ((const float4*)src)[i];
    v4h o = { (_Float16)v.x, (_Float16)v.y, (_Float16)v.z, (_Float16)v.w };
    ((v4h*)dst)[i] = o;
  }
}

// ---------------------------------------------------------------- fused QKV GEMM
// out = x @ W.T + b   (x: [4096,1024] f16, W: [1024,1024] f16 row-major)
// gridDim.z selects Q/K/V. Q,K stored [B,NH,S,HD]; V stored transposed [B,NH,HD,S].
__global__ __launch_bounds__(256) void qkv_gemm_kernel(
    const _Float16* __restrict__ xh,
    const _Float16* __restrict__ wqh, const _Float16* __restrict__ wkh,
    const _Float16* __restrict__ wvh,
    const float* __restrict__ bq, const float* __restrict__ bk,
    const float* __restrict__ bv,
    _Float16* __restrict__ qh, _Float16* __restrict__ kh,
    _Float16* __restrict__ vt) {
  int lane = threadIdx.x & 31;
  int wave = threadIdx.x >> 5;
  int z = blockIdx.z;
  const _Float16* W    = (z == 0) ? wqh : (z == 1) ? wkh : wvh;
  const float*    bias = (z == 0) ? bq  : (z == 1) ? bk  : bv;

  int m0 = blockIdx.x * 128 + wave * 16;   // 8 waves -> 128 rows per block
  int n0 = blockIdx.y * 64;

  v8f acc[4] = {};
  for (int k0 = 0; k0 < HIDc; k0 += 32) {
    if (k0 + 32 < HIDc)  // prefetch next A strip (emits global_prefetch_b8)
      __builtin_prefetch(xh + (size_t)(m0 + (lane & 15)) * HIDc + k0 + 32, 0, 3);
    v16h a = load_a_frag(xh, HIDc, m0, k0, lane);
#pragma unroll
    for (int nt = 0; nt < 4; ++nt) {
      v16h b = load_b_frag(W, HIDc, n0 + nt * 16, k0, lane);
      acc[nt] = __builtin_amdgcn_wmma_f32_16x16x32_f16(
          false, a, false, b, (short)0, acc[nt], false, false);
    }
  }

  int g = lane >> 4;
#pragma unroll
  for (int nt = 0; nt < 4; ++nt) {
    int col = n0 + nt * 16 + (lane & 15);
    float bval = bias[col];
    int h = col >> 6, d = col & 63;
#pragma unroll
    for (int j = 0; j < 8; ++j) {
      int row = m0 + j + 8 * g;
      int bi = row >> 10, si = row & 1023;
      float val = acc[nt][j] + bval;
      if (z < 2) {
        _Float16* dst = (z == 0) ? qh : kh;
        dst[(((size_t)(bi * NHc + h)) * Sc + si) * HDc + d] = (_Float16)val;
      } else {
        vt[(((size_t)(bi * NHc + h)) * HDc + d) * Sc + si] = (_Float16)val;
      }
    }
  }
}

// ---------------------------------------------------------------- flash attention
// grid: (S/64, B*NH), block: 128 (4 waves). Each wave: 16 query rows, full 64-dim acc.
__global__ __launch_bounds__(128) void flash_attn_kernel(
    const _Float16* __restrict__ qh, const _Float16* __restrict__ kh,
    const _Float16* __restrict__ vt, const float* __restrict__ mask,
    float* __restrict__ out) {
  __shared__ _Float16 plds[4 * 16 * 80];   // per-wave 16x64 prob tile, stride 80
  int lane = threadIdx.x & 31;
  int wave = threadIdx.x >> 5;
  int g = lane >> 4;
  int bh = blockIdx.y;             // b*NH + h
  int b = bh >> 4, h = bh & 15;
  int q0 = blockIdx.x * 64 + wave * 16;

  const _Float16* qbase = qh + (size_t)bh * Sc * HDc;
  const _Float16* kbase = kh + (size_t)bh * Sc * HDc;
  const _Float16* vbase = vt + (size_t)bh * HDc * Sc;

  v16h qa0 = load_a_frag(qbase, HDc, q0, 0, lane);
  v16h qa1 = load_a_frag(qbase, HDc, q0, 32, lane);

  float m[8], l[8];
#pragma unroll
  for (int j = 0; j < 8; ++j) { m[j] = -1.0e30f; l[j] = 0.0f; }
  v8f acc[4] = {};

  _Float16* myp = plds + wave * 16 * 80;

  for (int kt = 0; kt < Sc; kt += 64) {
    // ---- scores S = (Q K^T) * 1/sqrt(64) - (2-mask)*1e6, 16x64 tile ----
    v8f sv[4];
#pragma unroll
    for (int nt = 0; nt < 4; ++nt) {
      v16h kb0 = load_b_frag(kbase, HDc, kt + nt * 16, 0, lane);
      v16h kb1 = load_b_frag(kbase, HDc, kt + nt * 16, 32, lane);
      v8f c = {};
      c = __builtin_amdgcn_wmma_f32_16x16x32_f16(false, qa0, false, kb0, (short)0, c, false, false);
      c = __builtin_amdgcn_wmma_f32_16x16x32_f16(false, qa1, false, kb1, (short)0, c, false, false);
      int key = kt + nt * 16 + (lane & 15);            // column-constant per lane
      float mb = (2.0f - mask[b * Sc + key]) * 1.0e6f;
#pragma unroll
      for (int j = 0; j < 8; ++j) c[j] = c[j] * 0.125f - mb;
      sv[nt] = c;
    }

    // ---- online softmax (row = j + 8g, stats reduced across 16-lane group) ----
    float corr[8], rsum[8];
#pragma unroll
    for (int j = 0; j < 8; ++j) {
      float v = fmaxf(fmaxf(sv[0][j], sv[1][j]), fmaxf(sv[2][j], sv[3][j]));
#pragma unroll
      for (int off = 1; off < 16; off <<= 1) v = fmaxf(v, __shfl_xor(v, off, 32));
      float mn = fmaxf(m[j], v);
      corr[j] = __expf(m[j] - mn);
      m[j] = mn;
      rsum[j] = 0.0f;
    }
#pragma unroll
    for (int nt = 0; nt < 4; ++nt) {
#pragma unroll
      for (int j = 0; j < 8; ++j) {
        float p = __expf(sv[nt][j] - m[j]);
        rsum[j] += p;
        myp[(j + 8 * g) * 80 + nt * 16 + (lane & 15)] = (_Float16)p;  // C->LDS
      }
    }
#pragma unroll
    for (int j = 0; j < 8; ++j) {
      float v = rsum[j];
#pragma unroll
      for (int off = 1; off < 16; off <<= 1) v += __shfl_xor(v, off, 32);
      l[j] = l[j] * corr[j] + v;
    }
#pragma unroll
    for (int nt = 0; nt < 4; ++nt)
#pragma unroll
      for (int j = 0; j < 8; ++j) acc[nt][j] *= corr[j];

    // LDS write->read within same wave: drain DScnt, keep scheduler honest
    asm volatile("s_wait_dscnt 0" ::: "memory");
    __builtin_amdgcn_wave_barrier();

    // ---- P re-striped from LDS into A-fragment layout ----
    v16h pa0 = load_a_frag(myp, 80, 0, 0, lane);
    v16h pa1 = load_a_frag(myp, 80, 0, 32, lane);

    // ---- acc += P @ V  (V transposed: columns of B contiguous over keys) ----
#pragma unroll
    for (int nt = 0; nt < 4; ++nt) {
      v16h vb0 = load_b_frag(vbase, Sc, nt * 16, kt + 0, lane);
      v16h vb1 = load_b_frag(vbase, Sc, nt * 16, kt + 32, lane);
      acc[nt] = __builtin_amdgcn_wmma_f32_16x16x32_f16(false, pa0, false, vb0, (short)0, acc[nt], false, false);
      acc[nt] = __builtin_amdgcn_wmma_f32_16x16x32_f16(false, pa1, false, vb1, (short)0, acc[nt], false, false);
    }
  }

  // ---- epilogue: ctx[b, q, h*64+d] = acc / l ----
#pragma unroll
  for (int nt = 0; nt < 4; ++nt) {
    int d = nt * 16 + (lane & 15);
#pragma unroll
    for (int j = 0; j < 8; ++j) {
      int q = q0 + j + 8 * g;
      out[((size_t)b * Sc + q) * HIDc + h * HDc + d] = acc[nt][j] / l[j];
    }
  }
}

// ---------------------------------------------------------------- launch
extern "C" void kernel_launch(void* const* d_in, const int* in_sizes, int n_in,
                              void* d_out, int out_size, void* d_ws, size_t ws_size,
                              hipStream_t stream) {
  const float* hidden = (const float*)d_in[0];
  const float* mask   = (const float*)d_in[1];
  const float* Wq     = (const float*)d_in[2];
  const float* bq     = (const float*)d_in[3];
  const float* Wk     = (const float*)d_in[4];
  const float* bk     = (const float*)d_in[5];
  const float* Wv     = (const float*)d_in[6];
  const float* bv     = (const float*)d_in[7];
  float* out = (float*)d_out;

  char* ws = (char*)d_ws;
  _Float16* xh  = (_Float16*)(ws);                                   // 8 MB
  _Float16* wqh = (_Float16*)(ws + ( 8u << 20));                     // 2 MB
  _Float16* wkh = (_Float16*)(ws + (10u << 20));                     // 2 MB
  _Float16* wvh = (_Float16*)(ws + (12u << 20));                     // 2 MB
  _Float16* qh  = (_Float16*)(ws + (14u << 20));                     // 8 MB [B,NH,S,HD]
  _Float16* kh  = (_Float16*)(ws + (22u << 20));                     // 8 MB [B,NH,S,HD]
  _Float16* vt  = (_Float16*)(ws + (30u << 20));                     // 8 MB [B,NH,HD,S]

  // f32 -> f16 casts
  cast_f32_f16_kernel<<<(Mrows * HIDc / 4 + 255) / 256, 256, 0, stream>>>(hidden, xh, Mrows * HIDc / 4);
  cast_f32_f16_kernel<<<(HIDc * HIDc / 4 + 255) / 256, 256, 0, stream>>>(Wq, wqh, HIDc * HIDc / 4);
  cast_f32_f16_kernel<<<(HIDc * HIDc / 4 + 255) / 256, 256, 0, stream>>>(Wk, wkh, HIDc * HIDc / 4);
  cast_f32_f16_kernel<<<(HIDc * HIDc / 4 + 255) / 256, 256, 0, stream>>>(Wv, wvh, HIDc * HIDc / 4);

  // fused QKV projection: z selects Q/K/V
  qkv_gemm_kernel<<<dim3(Mrows / 128, HIDc / 64, 3), 256, 0, stream>>>(
      xh, wqh, wkh, wvh, bq, bk, bv, qh, kh, vt);

  // flash attention
  flash_attn_kernel<<<dim3(Sc / 64, Bc * NHc), 128, 0, stream>>>(qh, kh, vt, mask, out);
}
